// SelfAttention_22299470201668
// MI455X (gfx1250) — matrix operand order repaired and checked by reading, not compiled, not run on hardware
//
#include <hip/hip_runtime.h>

// ---------------- types ----------------
typedef __attribute__((ext_vector_type(16))) __bf16        v16bf;
typedef __attribute__((ext_vector_type(8)))  __bf16        v8bf;
typedef __attribute__((ext_vector_type(8)))  float         v8f;
typedef __attribute__((ext_vector_type(4)))  unsigned int  u32x4;
typedef __attribute__((ext_vector_type(8)))  unsigned short u16x8;

#define BDIM 2
#define SDIM 2048
#define DDIM 2048
#define HN   16
#define DH   128
#define MROWS (BDIM*SDIM)        // 4096
#define NORM_COEF 0.08838834764831845f  // 128^-0.5

// ---------------- helpers ----------------
__device__ __forceinline__ unsigned short f32_to_bf16(float f) {
    unsigned int u = __builtin_bit_cast(unsigned int, f);
    u += 0x7FFFu + ((u >> 16) & 1u);     // round-to-nearest-even
    return (unsigned short)(u >> 16);
}

__device__ __forceinline__ v8bf load8bf(const unsigned short* p) {
    u32x4 t = *reinterpret_cast<const u32x4*>(p);
    return __builtin_bit_cast(v8bf, t);
}

__device__ __forceinline__ v16bf cat16(v8bf lo, v8bf hi) {
    return __builtin_shufflevector(lo, hi, 0,1,2,3,4,5,6,7,8,9,10,11,12,13,14,15);
}

// A-matrix fragment (16x32 bf16, MxK): lane = row (mod 16);
// lanes 0-15 hold K {0..7,16..23}, lanes 16-31 hold K {8..15,24..31}.
__device__ __forceinline__ v16bf load_frag_a(const unsigned short* base, int stride, int lane) {
    int row  = lane & 15;
    int koff = (lane < 16) ? 0 : 8;
    const unsigned short* p = base + row * stride + koff;
    return cat16(load8bf(p), load8bf(p + 16));
}

// B-matrix fragment (32x16 bf16, KxN) sourced from a K-contiguous [N,K] array
// (i.e. computing X @ W^T): lane = column
// (lanes 0-15: K {0..7,16..23}, lanes 16-31: K {8..15,24..31}).
__device__ __forceinline__ v16bf load_frag_bt(const unsigned short* base, int stride, int lane) {
    int col  = lane & 15;
    int koff = (lane < 16) ? 0 : 8;
    const unsigned short* p = base + col * stride + koff;
    return cat16(load8bf(p), load8bf(p + 16));
}

// ---------------- f32 -> bf16 conversion ----------------
__global__ void cvt_bf16_kernel(const float* __restrict__ in, unsigned short* __restrict__ out, int n) {
    int i = blockIdx.x * blockDim.x + threadIdx.x;
    if (i < n) out[i] = f32_to_bf16(in[i]);
}

// ---------------- tiled WMMA GEMM:  C[M,N] = A[M,K] @ B[N,K]^T + bias ----------------
// mode 0: scatter bf16 into qkv layout [3][B][H][S][DH]; mode 1: f32 row-major out.
#define BM 128
#define BN 128
#define BK 64

__global__ __launch_bounds__(256)
void gemm_bf16_kernel(const unsigned short* __restrict__ A,
                      const unsigned short* __restrict__ Bm,
                      const float* __restrict__ bias,
                      int M, int N, int K, int mode,
                      unsigned short* __restrict__ out_qkv,
                      float* __restrict__ out_f32) {
    __shared__ alignas(16) unsigned short As[BM * BK];   // 16 KB
    __shared__ alignas(16) unsigned short Bs[BN * BK];   // 16 KB

    const int tid  = threadIdx.x;
    const int lane = tid & 31;
    const int wave = tid >> 5;               // 0..7
    const int wm   = (wave >> 1) * 32;       // 0,32,64,96
    const int wn   = (wave & 1) * 64;        // 0,64
    const int mbase = blockIdx.x * BM;
    const int nbase = blockIdx.y * BN;

    v8f acc[2][4];
    #pragma unroll
    for (int i = 0; i < 2; ++i)
        #pragma unroll
        for (int c = 0; c < 4; ++c)
            acc[i][c] = (v8f)0.0f;

    const int lr = tid >> 1;                 // 0..127 (tile row)
    const int lc = (tid & 1) * 32;           // 0 or 32 (tile col base, elems)

    for (int k0 = 0; k0 < K; k0 += BK) {
        // stage A and B tiles (each thread moves 4 x 16B per matrix)
        {
            const unsigned short* ga = A + (size_t)(mbase + lr) * K + k0 + lc;
            const unsigned short* gb = Bm + (size_t)(nbase + lr) * K + k0 + lc;
            #pragma unroll
            for (int q = 0; q < 4; ++q) {
                u32x4 a = *reinterpret_cast<const u32x4*>(ga + q * 8);
                *reinterpret_cast<u32x4*>(&As[lr * BK + lc + q * 8]) = a;
            }
            #pragma unroll
            for (int q = 0; q < 4; ++q) {
                u32x4 b = *reinterpret_cast<const u32x4*>(gb + q * 8);
                *reinterpret_cast<u32x4*>(&Bs[lr * BK + lc + q * 8]) = b;
            }
            if (k0 + BK < K) {               // prefetch next K-tile lines into caches
                __builtin_prefetch(ga + BK, 0, 0);
                __builtin_prefetch(gb + BK, 0, 0);
            }
        }
        __syncthreads();

        #pragma unroll
        for (int ks = 0; ks < BK / 32; ++ks) {
            v16bf af0 = load_frag_a(&As[(wm + 0)  * BK + ks * 32], BK, lane);
            v16bf af1 = load_frag_a(&As[(wm + 16) * BK + ks * 32], BK, lane);
            v16bf bf[4];
            #pragma unroll
            for (int c = 0; c < 4; ++c)
                bf[c] = load_frag_bt(&Bs[(wn + c * 16) * BK + ks * 32], BK, lane);
            #pragma unroll
            for (int c = 0; c < 4; ++c) {
                acc[0][c] = __builtin_amdgcn_wmma_f32_16x16x32_bf16(false, af0, false, bf[c],
                                                                    (short)0, acc[0][c], false, false);
                acc[1][c] = __builtin_amdgcn_wmma_f32_16x16x32_bf16(false, af1, false, bf[c],
                                                                    (short)0, acc[1][c], false, false);
            }
        }
        __syncthreads();
    }

    // epilogue: lane owns one column per tile, 8 rows per accumulator
    const int colLocal = lane & 15;
    #pragma unroll
    for (int i = 0; i < 2; ++i) {
        #pragma unroll
        for (int c = 0; c < 4; ++c) {
            const int n  = nbase + wn + c * 16 + colLocal;
            const float bv = bias[n];
            #pragma unroll
            for (int r = 0; r < 8; ++r) {
                const int m = mbase + wm + i * 16 + ((lane < 16) ? r : (8 + r));
                const float val = acc[i][c][r] + bv;
                if (mode == 0) {
                    // n -> (which, h, dh), m -> (b, s); sizes are powers of two
                    const int which = n >> 11;
                    const int h  = (n >> 7) & 15;
                    const int dh = n & 127;
                    const int b  = m >> 11;
                    const int s  = m & 2047;
                    const size_t dst = ((((size_t)which * BDIM + b) * HN + h) * SDIM + s) * DH + dh;
                    out_qkv[dst] = f32_to_bf16(val);
                } else {
                    out_f32[(size_t)m * N + n] = val;
                }
            }
        }
    }
}

// ---------------- flash attention over 64-row q-tiles ----------------
// qkv: bf16 [3][B][H][S][DH]; mask: f32 [B,1,S,S]; ctx: bf16 [B*S, D] (heads merged)
__global__ __launch_bounds__(128)
void attn_kernel(const unsigned short* __restrict__ qkv,
                 const float* __restrict__ mask,
                 unsigned short* __restrict__ ctx) {
    __shared__ alignas(16) unsigned short Qs[64 * DH];       // 16 KB
    __shared__ alignas(16) unsigned short Ks[64 * DH];       // 16 KB
    __shared__ alignas(16) unsigned short Vt[DH * 64];       // 16 KB, transposed V
    __shared__ alignas(16) unsigned short Ps[4][16 * 64];    // per-wave P tiles

    const int tid  = threadIdx.x;
    const int lane = tid & 31;
    const int wave = tid >> 5;               // 0..3
    const int blk  = blockIdx.x;
    const int qt   = blk & 31;               // S/64 = 32
    const int h    = (blk >> 5) & 15;
    const int b    = blk >> 9;
    const int qbase = qt * 64;

    const size_t headStride = (size_t)SDIM * DH;
    const unsigned short* Qg = qkv + (((size_t)0 * BDIM + b) * HN + h) * headStride;
    const unsigned short* Kg = qkv + (((size_t)1 * BDIM + b) * HN + h) * headStride;
    const unsigned short* Vg = qkv + (((size_t)2 * BDIM + b) * HN + h) * headStride;

    // stage this block's Q tile once
    for (int i = tid; i < 64 * DH / 8; i += 128) {
        u32x4 v = *reinterpret_cast<const u32x4*>(Qg + (size_t)qbase * DH + i * 8);
        *reinterpret_cast<u32x4*>(&Qs[i * 8]) = v;
    }

    float mrow[8], lrow[8];
    v8f octx[8];
    #pragma unroll
    for (int r = 0; r < 8; ++r) { mrow[r] = -3.0e38f; lrow[r] = 0.0f; }
    #pragma unroll
    for (int dt = 0; dt < 8; ++dt) octx[dt] = (v8f)0.0f;

    for (int kt = 0; kt < SDIM / 64; ++kt) {
        const int kb = kt * 64;
        __syncthreads();
        // stage K tile (row major) and V tile (transposed -> Vt[d][k])
        for (int i = tid; i < 1024; i += 128) {
            u32x4 v = *reinterpret_cast<const u32x4*>(Kg + (size_t)kb * DH + i * 8);
            *reinterpret_cast<u32x4*>(&Ks[i * 8]) = v;
            if (kt + 1 < SDIM / 64)
                __builtin_prefetch(Kg + (size_t)(kb + 64) * DH + i * 8, 0, 0);
        }
        for (int i = tid; i < 1024; i += 128) {
            const int elem = i * 8;
            const int krow = elem >> 7;      // /DH
            const int d0   = elem & 127;
            u16x8 v = *reinterpret_cast<const u16x8*>(Vg + (size_t)(kb + krow) * DH + d0);
            #pragma unroll
            for (int j = 0; j < 8; ++j) Vt[(d0 + j) * 64 + krow] = v[j];
            if (kt + 1 < SDIM / 64)
                __builtin_prefetch(Vg + (size_t)(kb + 64 + krow) * DH + d0, 0, 0);
        }
        __syncthreads();

        // scores S = Q @ K^T for this wave's 16 rows x 64 keys
        v8f sacc[4];
        #pragma unroll
        for (int c = 0; c < 4; ++c) sacc[c] = (v8f)0.0f;
        #pragma unroll
        for (int kk = 0; kk < 4; ++kk) {
            v16bf aq = load_frag_a(&Qs[(wave * 16) * DH + kk * 32], DH, lane);
            #pragma unroll
            for (int c = 0; c < 4; ++c) {
                v16bf bk = load_frag_bt(&Ks[(c * 16) * DH + kk * 32], DH, lane);
                sacc[c] = __builtin_amdgcn_wmma_f32_16x16x32_bf16(false, aq, false, bk,
                                                                  (short)0, sacc[c], false, false);
            }
        }

        // scale + additive mask, running row max
        float sval[4][8];
        float pmax[8];
        #pragma unroll
        for (int r = 0; r < 8; ++r) pmax[r] = -3.0e38f;
        #pragma unroll
        for (int c = 0; c < 4; ++c) {
            const int kcol = kb + c * 16 + (lane & 15);
            #pragma unroll
            for (int r = 0; r < 8; ++r) {
                const int qrow = qbase + wave * 16 + ((lane < 16) ? r : (8 + r));
                float s = sacc[c][r] * NORM_COEF
                        + mask[((size_t)b * SDIM + qrow) * SDIM + kcol];
                sval[c][r] = s;
                pmax[r] = fmaxf(pmax[r], s);
            }
        }
        // reduce row max across the 16-lane group owning this tile's columns
        #pragma unroll
        for (int off = 8; off >= 1; off >>= 1)
            #pragma unroll
            for (int r = 0; r < 8; ++r)
                pmax[r] = fmaxf(pmax[r], __shfl_xor(pmax[r], off, 32));

        float alpha[8], mnew[8];
        #pragma unroll
        for (int r = 0; r < 8; ++r) {
            mnew[r]  = fmaxf(mrow[r], pmax[r]);
            alpha[r] = __expf(mrow[r] - mnew[r]);
            mrow[r]  = mnew[r];
        }

        // p = exp(s - m), stash bf16 P into this wave's LDS tile, partial row sums
        float psum[8];
        #pragma unroll
        for (int r = 0; r < 8; ++r) psum[r] = 0.0f;
        #pragma unroll
        for (int c = 0; c < 4; ++c) {
            #pragma unroll
            for (int r = 0; r < 8; ++r) {
                const int mloc = (lane < 16) ? r : (8 + r);
                float p = __expf(sval[c][r] - mnew[r]);
                psum[r] += p;
                Ps[wave][mloc * 64 + c * 16 + (lane & 15)] = f32_to_bf16(p);
            }
        }
        #pragma unroll
        for (int off = 8; off >= 1; off >>= 1)
            #pragma unroll
            for (int r = 0; r < 8; ++r)
                psum[r] += __shfl_xor(psum[r], off, 32);
        #pragma unroll
        for (int r = 0; r < 8; ++r) lrow[r] = lrow[r] * alpha[r] + psum[r];
        #pragma unroll
        for (int dt = 0; dt < 8; ++dt)
            #pragma unroll
            for (int r = 0; r < 8; ++r)
                octx[dt][r] = octx[dt][r] * alpha[r];

        __syncthreads();   // make this wave's Ps writes visible to all its lanes

        // ctx += P @ V  (A from Ps, B from transposed V)
        v16bf ap0 = load_frag_a(&Ps[wave][0],  64, lane);
        v16bf ap1 = load_frag_a(&Ps[wave][32], 64, lane);
        #pragma unroll
        for (int dt = 0; dt < 8; ++dt) {
            v16bf bv0 = load_frag_bt(&Vt[(dt * 16) * 64 + 0],  64, lane);
            v16bf bv1 = load_frag_bt(&Vt[(dt * 16) * 64 + 32], 64, lane);
            octx[dt] = __builtin_amdgcn_wmma_f32_16x16x32_bf16(false, ap0, false, bv0,
                                                               (short)0, octx[dt], false, false);
            octx[dt] = __builtin_amdgcn_wmma_f32_16x16x32_bf16(false, ap1, false, bv1,
                                                               (short)0, octx[dt], false, false);
        }
    }

    // normalize and emit merged-head bf16 ctx [B*S, D]
    float inv[8];
    #pragma unroll
    for (int r = 0; r < 8; ++r) inv[r] = (lrow[r] > 0.0f) ? (1.0f / lrow[r]) : 0.0f;
    #pragma unroll
    for (int dt = 0; dt < 8; ++dt) {
        const int d = dt * 16 + (lane & 15);
        #pragma unroll
        for (int r = 0; r < 8; ++r) {
            const int qrow = qbase + wave * 16 + ((lane < 16) ? r : (8 + r));
            ctx[((size_t)b * SDIM + qrow) * DDIM + h * DH + d] = f32_to_bf16(octx[dt][r] * inv[r]);
        }
    }
}

// ---------------- host orchestration ----------------
extern "C" void kernel_launch(void* const* d_in, const int* in_sizes, int n_in,
                              void* d_out, int out_size, void* d_ws, size_t ws_size,
                              hipStream_t stream) {
    const float* hidden = (const float*)d_in[0];   // [B,S,D]
    const float* mask   = (const float*)d_in[1];   // [B,1,S,S]
    const float* Wqkv   = (const float*)d_in[2];   // [3D,D]
    const float* bqkv   = (const float*)d_in[3];   // [3D]
    const float* Wd     = (const float*)d_in[4];   // [D,D]
    const float* bd     = (const float*)d_in[5];   // [D]
    float* out = (float*)d_out;

    char* ws = (char*)d_ws;
    size_t off = 0;
    unsigned short* Xb     = (unsigned short*)(ws + off); off += (size_t)MROWS * DDIM * 2;       // 16 MB
    unsigned short* Wqkvb  = (unsigned short*)(ws + off); off += (size_t)3 * DDIM * DDIM * 2;    // 24 MB
    unsigned short* Wdb    = (unsigned short*)(ws + off); off += (size_t)DDIM * DDIM * 2;        //  8 MB
    unsigned short* QKVb   = (unsigned short*)(ws + off); off += (size_t)3 * MROWS * DDIM * 2;   // 48 MB
    unsigned short* CTXb   = (unsigned short*)(ws + off); off += (size_t)MROWS * DDIM * 2;       // 16 MB
    (void)ws_size; (void)in_sizes; (void)n_in; (void)out_size;

    // f32 -> bf16 staging
    {
        const int nX = MROWS * DDIM;             // 8,388,608
        const int nW = 3 * DDIM * DDIM;          // 12,582,912
        const int nD = DDIM * DDIM;              // 4,194,304
        cvt_bf16_kernel<<<(nX + 255) / 256, 256, 0, stream>>>(hidden, Xb, nX);
        cvt_bf16_kernel<<<(nW + 255) / 256, 256, 0, stream>>>(Wqkv, Wqkvb, nW);
        cvt_bf16_kernel<<<(nD + 255) / 256, 256, 0, stream>>>(Wd, Wdb, nD);
    }

    // fused QKV projection: [4096, 2048] @ [6144, 2048]^T + b  -> qkv layout
    gemm_bf16_kernel<<<dim3(MROWS / BM, (3 * DDIM) / BN), 256, 0, stream>>>(
        Xb, Wqkvb, bqkv, MROWS, 3 * DDIM, DDIM, /*mode=*/0, QKVb, nullptr);

    // attention: one block per (b, h, 64-row q tile)
    attn_kernel<<<dim3(BDIM * HN * (SDIM / 64)), 128, 0, stream>>>(QKVb, mask, CTXb);

    // dense projection: [4096, 2048] @ [2048, 2048]^T + b -> f32 out
    gemm_bf16_kernel<<<dim3(MROWS / BM, DDIM / BN), 256, 0, stream>>>(
        CTXb, Wdb, bd, MROWS, DDIM, DDIM, /*mode=*/1, nullptr, out);
}
